// StreamGTCRNNoScatter_53833120088181
// MI455X (gfx1250) — matrix operand order repaired and checked by reading, not compiled, not run on hardware
//
#include <hip/hip_runtime.h>

// ============================================================================
// StreamGTCRN fused megakernel for gfx1250 (MI455X, wave32, WMMA + TDM).
// One workgroup (128 threads = 4 wave32) per batch element; all activations
// live in LDS; HBM is touched only for spec, recurrent caches, params, outputs.
// Dense ops use v_wmma_f32_16x16x32_f16; conv-cache tiles are pulled into LDS
// with tensor_load_to_lds (TDM) overlapped with the pc1 WMMA.
// ============================================================================

typedef __attribute__((ext_vector_type(16))) _Float16 v16h;
typedef __attribute__((ext_vector_type(8)))  float    v8f;
typedef __attribute__((ext_vector_type(4)))  unsigned v4u;
typedef __attribute__((ext_vector_type(8)))  int      v8i;
typedef __attribute__((ext_vector_type(4)))  int      v4i;

#define NT_BLK 128

#if __has_builtin(__builtin_amdgcn_tensor_load_to_lds)
#define HAVE_TDM 1
#else
#define HAVE_TDM 0
#endif

__device__ __forceinline__ float sigm(float x) { return 1.0f / (1.0f + __expf(-x)); }

#if HAVE_TDM
// ---------------------------------------------------------------------------
// TDM 2D tile load: nrows rows of rowlen f32 elements, global row stride
// rowstride (elements) -> contiguous LDS at ldst. D# per CDNA5 ISA 8.3/8.4:
// group0: count=1 | lds_addr | global_addr[56:0] | type=2
// group1: data_size=4B, tensor_dim0=rowlen, tensor_dim1=nrows,
//         tile_dim0=rowlen, tile_dim1=nrows, tensor_dim0_stride=rowstride
// Issue from one wave; completion via s_wait_tensorcnt.
// ---------------------------------------------------------------------------
__device__ __forceinline__ void tdm_load_2d(const float* gsrc, float* ldst,
                                            unsigned rowlen, unsigned nrows,
                                            unsigned rowstride)
{
  const unsigned long long ga = (unsigned long long)gsrc;
  const unsigned la = (unsigned)(unsigned long long)(void*)ldst;
  v4u g0;
  g0.x = 1u;                                                  // count = 1
  g0.y = la;                                                  // lds_addr (bytes)
  g0.z = (unsigned)(ga & 0xffffffffull);                      // global_addr[31:0]
  g0.w = (unsigned)((ga >> 32) & 0x1ffffffull) | (2u << 30);  // [56:32] | type=2
  v8i g1;
  g1[0] = (int)(2u << 16);                                    // data_size = 4B
  g1[1] = (int)((rowlen & 0xffffu) << 16);                    // tensor_dim0 lo
  g1[2] = (int)(((rowlen >> 16) & 0xffffu) | ((nrows & 0xffffu) << 16)); // dim0 hi|dim1 lo
  g1[3] = (int)(((nrows >> 16) & 0xffffu) | ((rowlen & 0xffffu) << 16)); // dim1 hi|tile_dim0
  g1[4] = (int)(nrows & 0xffffu);                             // tile_dim1 (tile_dim2=0)
  g1[5] = (int)rowstride;                                     // tensor_dim0_stride lo
  g1[6] = 0;
  g1[7] = 0;
  v4i gz = {0, 0, 0, 0};
#if defined(__clang_major__) && (__clang_major__ >= 23)
  v8i gz8 = {0, 0, 0, 0, 0, 0, 0, 0};
  __builtin_amdgcn_tensor_load_to_lds(g0, g1, gz, gz, gz8, 0);
#else
  __builtin_amdgcn_tensor_load_to_lds(g0, g1, gz, gz, 0);
#endif
}
#endif // HAVE_TDM

// ---------------------------------------------------------------------------
// C[M][N] = A[M][K] * B[K][N], f16 WMMA with f32 accumulate.
// A[m][k] = at ? A[k*lda+m] : A[m*lda+k]        (LDS, f32)
// B is a weight matrix stored (N,K) row-major:  B[k][n] = Bw[n*ldb + k]
// Branchless: clamped-address loads + cndmask zero-select; out-of-range C
// lanes store to an LDS dump slot (address select, unconditional store).
// Per-lane packing: A and B both use contiguous k = 16*half + j (not the ISA table order).
// ---------------------------------------------------------------------------
__device__ void block_gemm_f16(const float* A, int lda, bool at,
                               const float* Bw, int ldb,
                               float* C, int ldc, int M, int N, int K)
{
  __shared__ float dump[32];
  const int lane = threadIdx.x & 31;
  const int wave = threadIdx.x >> 5;
  const int nw   = blockDim.x >> 5;
  const int half = lane >> 4;
  const int l16  = lane & 15;
  const int mt = (M + 15) >> 4;
  const int nt = (N + 15) >> 4;
  const int Mc = M - 1, Nc = N - 1, Kc = K - 1;
  for (int t = wave; t < mt * nt; t += nw) {
    const int m0 = (t / nt) << 4;
    const int n0 = (t % nt) << 4;
    const int am = m0 + l16;
    const int bn = n0 + l16;
    const int amc = am <= Mc ? am : Mc;
    const int bnc = bn <= Nc ? bn : Nc;
    const bool aok = am < M;
    const bool bok = bn < N;
    v8f acc = {};
    for (int k0 = 0; k0 < K; k0 += 32) {
      v16h av, bv;
      const int kbase = k0 + half * 16;   // B: lanes 0-15 K=k0..k0+15, 16-31 +16
      if (bok && (kbase + 16) <= K) {
        // contiguous row segment -> 4x global_load_b128
        const float4* bp = (const float4*)(Bw + (size_t)bnc * ldb + kbase);
#pragma unroll
        for (int q = 0; q < 4; ++q) {
          const float4 v = bp[q];
          bv[q * 4 + 0] = (_Float16)v.x; bv[q * 4 + 1] = (_Float16)v.y;
          bv[q * 4 + 2] = (_Float16)v.z; bv[q * 4 + 3] = (_Float16)v.w;
        }
      } else {
#pragma unroll
        for (int j = 0; j < 16; ++j) {
          const int kb = kbase + j;
          const float v = Bw[(size_t)bnc * ldb + (kb <= Kc ? kb : Kc)];
          bv[j] = (_Float16)((bok && kb < K) ? v : 0.0f);
        }
      }
#pragma unroll
      for (int j = 0; j < 16; ++j) {
        // A 16x32 f16: K = 16*half + j
        const int ka = k0 + half * 16 + j;
        const int kac = ka <= Kc ? ka : Kc;
        const float v = at ? A[kac * lda + amc] : A[amc * lda + kac];
        av[j] = (_Float16)((aok && ka < K) ? v : 0.0f);
      }
      acc = __builtin_amdgcn_wmma_f32_16x16x32_f16(false, av, false, bv,
                                                   (short)0, acc, false, false);
    }
    const int cn = n0 + l16;
#pragma unroll
    for (int r = 0; r < 8; ++r) {
      const int cm = m0 + r + (half << 3);   // C/D: vgpr r -> M = r + 8*half
      float* cp = (cm < M && cn < N) ? (C + cm * ldc + cn) : (dump + lane);
      *cp = acc[r];
    }
  }
}

__device__ __forceinline__ void block_reduce2(float& a, float& b, float* s_red)
{
  const int tid = threadIdx.x, NT = blockDim.x;
  s_red[tid] = a; s_red[NT + tid] = b;
  __syncthreads();
  for (int off = NT >> 1; off > 0; off >>= 1) {
    if (tid < off) { s_red[tid] += s_red[tid + off]; s_red[NT + tid] += s_red[NT + tid + off]; }
    __syncthreads();
  }
  a = s_red[0]; b = s_red[NT];
  __syncthreads();
}

// ---------------- parameter pointer tables -----------------
struct BNp { const float *b, *g, *m, *v; };
struct CBp { const float *a, *b; BNp bn; const float *w; };
struct GTp { const float *a1, *a2; BNp bn1, bn2, bn3;
             const float *dw_b, *dw_w, *fc_b, *fc_w, *gWh, *gWi, *gbh, *gbi,
                         *pc1_b, *pc1_w, *pc2_b, *pc2_w; };
struct DPp { const float *iaWh[2][2], *iaWi[2][2], *iabh[2][2], *iabi[2][2]; // [group][0=_b,1=_f]
             const float *ieWh[2], *ieWi[2], *iebh[2], *iebi[2];
             const float *inter_fc_b, *inter_fc_w, *inter_ln_b, *inter_ln_g;
             const float *intra_fc_b, *intra_fc_w, *intra_ln_b, *intra_ln_g; };
struct MPa { GTp de0, de1, de2; CBp de3, de4; DPp dp1, dp2; CBp en0, en1;
             GTp en2, en3, en4; const float *erb_bm, *erb_bs; };
struct KArgs {
  const float *spec;
  const float *ecc_in[3], *dcc_in[3], *etc_in[3], *dtc_in[3], *ic_in[2];
  float *o_spec;
  float *ecc_out[3], *dcc_out[3], *etc_out[3], *dtc_out[3], *ic_out[2];
  MPa P;
};

// ---------------------------------------------------------------------------
// GT block: TDM loads of cache rows (t=0, t=dil) overlapped with pc1 WMMA ->
// dilated depthwise 3x3 -> pc2 WMMA -> temporal-attention GRU(8->16) + FC ->
// channel interleave. Also shifts the conv cache (global->global).
// ---------------------------------------------------------------------------
__device__ __noinline__ void gt_block(const GTp& P, const float* xin, float* xout,
                                      const float* cin, float* cout,
                                      const float* tin, float* tout,
                                      int dil, int deconv,
                                      float* s_h, float* s_c0, float* s_cd,
                                      float* s_y, float* s_y2, float* s_mn,
                                      float* s_zt, float* s_gg, float* s_htra, float* s_at)
{
  const int tid = threadIdx.x;
  const int NT = blockDim.x;
  const int T2 = 2 * dil;

#if HAVE_TDM
  // kick off DMA of the two needed cache time-rows while pc1 runs on WMMA
  if (tid < 32) {
    tdm_load_2d(cin, s_c0, 33u, 16u, (unsigned)(T2 * 33));
    tdm_load_2d(cin + (size_t)dil * 33, s_cd, 33u, 16u, (unsigned)(T2 * 33));
  }
#endif

  // pc1: (33x8) @ (16x8)^T   A = x1 in (C,F) layout -> at=true
  block_gemm_f16(xin, 33, true, P.pc1_w, 8, s_mn, 16, 33, 16, 8);
  __syncthreads();
  {
    const float a1 = P.a1[0];
    for (int i = tid; i < 16 * 33; i += NT) {
      const int c = i / 33, f = i % 33;
      float v = s_mn[f * 16 + c] + P.pc1_b[c];
      v = (v - P.bn1.m[c]) * (P.bn1.g[c] * rsqrtf(P.bn1.v[c] + 1e-5f)) + P.bn1.b[c];
      s_h[i] = v >= 0.f ? v : a1 * v;
#if !HAVE_TDM
      s_c0[i] = cin[(c * T2 + 0) * 33 + f];
      s_cd[i] = cin[(c * T2 + dil) * 33 + f];
#endif
    }
  }
#if HAVE_TDM
  if (tid < 32) __builtin_amdgcn_s_wait_tensorcnt(0);
#endif
  __syncthreads();
  // depthwise 3x3 (dilated in time; freq pad 1) + bn2 + prelu
  {
    const float a2 = P.a2[0];
    for (int i = tid; i < 16 * 33; i += NT) {
      const int c = i / 33, f = i % 33;
      float acc = P.dw_b[c];
#pragma unroll
      for (int kt = 0; kt < 3; ++kt) {
        const float* row = (kt == 0) ? s_c0 : (kt == 1) ? s_cd : s_h;
#pragma unroll
        for (int kf = 0; kf < 3; ++kf) {
          const int ff = f + kf - 1;
          if (ff < 0 || ff > 32) continue;
          const int wkt = deconv ? 2 - kt : kt;
          const int wkf = deconv ? 2 - kf : kf;
          acc += P.dw_w[c * 9 + wkt * 3 + wkf] * row[c * 33 + ff];
        }
      }
      acc = (acc - P.bn2.m[c]) * (P.bn2.g[c] * rsqrtf(P.bn2.v[c] + 1e-5f)) + P.bn2.b[c];
      s_y[i] = acc >= 0.f ? acc : a2 * acc;
    }
  }
  // shift conv cache (streamed global->global)
  for (int i = tid; i < 16 * T2 * 33; i += NT) {
    const int f = i % 33;
    const int t = (i / 33) % T2;
    const int c = i / (33 * T2);
    const float v = (t < T2 - 1) ? cin[(c * T2 + t + 1) * 33 + f] : s_h[c * 33 + f];
    cout[(c * T2 + t) * 33 + f] = v;
  }
  __syncthreads();
  // pc2: (33x16) @ (8x16)^T
  block_gemm_f16(s_y, 33, true, P.pc2_w, 16, s_mn, 8, 33, 8, 16);
  __syncthreads();
  for (int i = tid; i < 8 * 33; i += NT) {
    const int c = i / 33, f = i % 33;
    float v = s_mn[f * 8 + c] + P.pc2_b[c];
    v = (v - P.bn3.m[c]) * (P.bn3.g[c] * rsqrtf(P.bn3.v[c] + 1e-5f)) + P.bn3.b[c];
    s_y2[i] = v;
  }
  __syncthreads();
  if (tid < 8) {
    float s = 0.f;
    for (int f = 0; f < 33; ++f) { const float v = s_y2[tid * 33 + f]; s += v * v; }
    s_zt[tid] = s * (1.f / 33.f);
  }
  __syncthreads();
  // GRU(8 -> 16): 48 gates, one thread each
  if (tid < 48) {
    float gi = P.gbi[tid], gh = P.gbh[tid];
    for (int k = 0; k < 8; ++k)  gi += s_zt[k] * P.gWi[tid * 8 + k];
    for (int k = 0; k < 16; ++k) gh += tin[k] * P.gWh[tid * 16 + k];
    s_gg[tid] = gi; s_gg[48 + tid] = gh;
  }
  __syncthreads();
  if (tid < 16) {
    const float r = sigm(s_gg[tid] + s_gg[48 + tid]);
    const float z = sigm(s_gg[16 + tid] + s_gg[64 + tid]);
    const float n = tanhf(s_gg[32 + tid] + r * s_gg[80 + tid]);
    const float h = (1.f - z) * n + z * tin[tid];
    s_htra[tid] = h;
    tout[tid] = h;
  }
  __syncthreads();
  if (tid < 8) {
    float a = P.fc_b[tid];
    for (int k = 0; k < 16; ++k) a += s_htra[k] * P.fc_w[tid * 16 + k];
    s_at[tid] = sigm(a);
  }
  __syncthreads();
  // out channels interleave: [y0*at0, x2_0, y1*at1, x2_1, ...]
  for (int i = tid; i < 16 * 33; i += NT) {
    const int c = i / 33, f = i % 33;
    xout[i] = (c & 1) ? xin[(8 + (c >> 1)) * 33 + f]
                      : s_y2[(c >> 1) * 33 + f] * s_at[c >> 1];
  }
  __syncthreads();
}

// ---------------------------------------------------------------------------
// Dual-path GRNN: 4 tiny intra-GRU scans (one per wave32) + WMMA FC/LN +
// per-frequency inter GRU (8->8 x2) with WMMA gate GEMMs.
// ---------------------------------------------------------------------------
__device__ __noinline__ void dpgrnn(const DPp& P, const float* xin, float* xout,
                                    const float* ic_in, float* ic_out,
                                    float* s_i16, float* s_g1, float* s_g2,
                                    float* s_io, float* s_y16, float* s_hc, float* s_red)
{
  const int tid = threadIdx.x;
  const int NT = blockDim.x;
  const int wave = tid >> 5;
  const int lane = tid & 31;

  // ---- intra: 4 independent GRU(8->4) scans over F, one per wave ----
  if (wave < 4) {
    const int g = wave >> 1;
    const int bwd = wave & 1;
    const int sel = bwd ? 0 : 1;       // sorted leaves: [0]='_b', [1]='_f'
    const float* Wi = P.iaWi[g][sel];
    const float* Wh = P.iaWh[g][sel];
    const float* bi = P.iabi[g][sel];
    const float* bh = P.iabh[g][sel];
    float hv[4] = {0.f, 0.f, 0.f, 0.f};
    for (int s = 0; s < 33; ++s) {
      const int f = bwd ? (32 - s) : s;
      float xv[8];
#pragma unroll
      for (int c = 0; c < 8; ++c) xv[c] = xin[(g * 8 + c) * 33 + f];
      float gi[12], gh[12];
#pragma unroll
      for (int j = 0; j < 12; ++j) {
        float a = bi[j], b2 = bh[j];
#pragma unroll
        for (int c = 0; c < 8; ++c) a += xv[c] * Wi[j * 8 + c];
#pragma unroll
        for (int c = 0; c < 4; ++c) b2 += hv[c] * Wh[j * 4 + c];
        gi[j] = a; gh[j] = b2;
      }
#pragma unroll
      for (int c = 0; c < 4; ++c) {
        const float r = sigm(gi[c] + gh[c]);
        const float z = sigm(gi[4 + c] + gh[4 + c]);
        const float n = tanhf(gi[8 + c] + r * gh[8 + c]);
        hv[c] = (1.f - z) * n + z * hv[c];
      }
      if (lane < 4) s_i16[f * 16 + g * 8 + bwd * 4 + lane] = hv[lane];
    }
  }
  __syncthreads();

  // intra_fc: (33x16) @ (16x16)^T
  block_gemm_f16(s_i16, 16, false, P.intra_fc_w, 16, s_g1, 16, 33, 16, 16);
  __syncthreads();
  float ps = 0.f, pq = 0.f;
  for (int i = tid; i < 528; i += NT) {
    const float v = s_g1[i] + P.intra_fc_b[i & 15];
    s_g1[i] = v; ps += v; pq += v * v;
  }
  block_reduce2(ps, pq, s_red);
  {
    const float mean = ps / 528.f;
    const float var = pq / 528.f - mean * mean;
    const float rstd = rsqrtf(var + 1e-8f);
    for (int i = tid; i < 528; i += NT) {
      const int f = i >> 4, c = i & 15;
      const float v = (s_g1[i] - mean) * rstd * P.intra_ln_g[i] + P.intra_ln_b[i];
      s_io[i] = xin[c * 33 + f] + v;
    }
  }
  for (int i = tid; i < 528; i += NT) s_hc[i] = ic_in[i];
  __syncthreads();

  // ---- inter: two GRU(8->8) over 33 frequency rows ----
  for (int g = 0; g < 2; ++g) {
    block_gemm_f16(s_io + g * 8, 16, false, P.ieWi[g], 8, s_g1, 24, 33, 24, 8);
    block_gemm_f16(s_hc + g * 8, 16, false, P.ieWh[g], 8, s_g2, 24, 33, 24, 8);
    __syncthreads();
    const float* bi2 = P.iebi[g];
    const float* bh2 = P.iebh[g];
    for (int i = tid; i < 33 * 8; i += NT) {
      const int f = i >> 3, c = i & 7;
      const float ir = s_g1[f * 24 + c] + bi2[c];
      const float iz = s_g1[f * 24 + 8 + c] + bi2[8 + c];
      const float in_ = s_g1[f * 24 + 16 + c] + bi2[16 + c];
      const float hr = s_g2[f * 24 + c] + bh2[c];
      const float hz = s_g2[f * 24 + 8 + c] + bh2[8 + c];
      const float hn = s_g2[f * 24 + 16 + c] + bh2[16 + c];
      const float r = sigm(ir + hr);
      const float z = sigm(iz + hz);
      const float n = tanhf(in_ + r * hn);
      const float hp = s_hc[f * 16 + g * 8 + c];
      s_y16[f * 16 + g * 8 + c] = (1.f - z) * n + z * hp;
    }
    __syncthreads();
  }
  for (int i = tid; i < 528; i += NT) ic_out[i] = s_y16[i];

  // inter_fc + LN + residual -> (C,F)
  block_gemm_f16(s_y16, 16, false, P.inter_fc_w, 16, s_g1, 16, 33, 16, 16);
  __syncthreads();
  ps = 0.f; pq = 0.f;
  for (int i = tid; i < 528; i += NT) {
    const float v = s_g1[i] + P.inter_fc_b[i & 15];
    s_g1[i] = v; ps += v; pq += v * v;
  }
  block_reduce2(ps, pq, s_red);
  {
    const float mean = ps / 528.f;
    const float var = pq / 528.f - mean * mean;
    const float rstd = rsqrtf(var + 1e-8f);
    for (int i = tid; i < 528; i += NT) {
      const int f = i >> 4, c = i & 15;
      const float v = (s_g1[i] - mean) * rstd * P.inter_ln_g[i] + P.inter_ln_b[i];
      xout[c * 33 + f] = s_io[i] + v;
    }
  }
  __syncthreads();
}

// ---------------------------------------------------------------------------
__global__ __launch_bounds__(NT_BLK) void gtcrn_mega_kernel(KArgs A)
{
  __shared__ float s_fh[771];                // mag / sr / si  (3 x 257)
  __shared__ float s_fe[387];                // ERB feature    (3 x 129)
  __shared__ float s_sfe[9 * 133];           // SFE, 2-col halo each side
  __shared__ float s_e0[16 * 69];            // 16 x 65 skip, 2-col halo
  __shared__ float s_e1[528], s_e2[528], s_e3[528], s_e4[528];
  __shared__ float s_xa[528], s_xb[528];     // ping-pong activations (16 x 33)
  __shared__ float s_h[528], s_c0[528], s_cd[528], s_y[528];
  __shared__ float s_y2[264];
  __shared__ float s_g1[792], s_g2[792];     // GEMM staging (33 x 24 max)
  __shared__ float s_io[528], s_i16[528], s_y16[528], s_hc[528];
  __shared__ float s_x65[1040];
  __shared__ float s_d4[258];                // 2 x 129
  __shared__ float s_mE[384];                // 2 x 192 ERB synthesis
  __shared__ float s_zt[8], s_gg[96], s_htra[16], s_at[8];
  __shared__ float s_red[2 * NT_BLK];

  const int b = blockIdx.x;
  const int tid = threadIdx.x;
  const int NT = blockDim.x;
  const MPa& P = A.P;
  const size_t b16 = (size_t)b * 16;

  // warm L2/L0 for the big recurrent caches (global_prefetch_b8)
  if (tid < 64) {
    __builtin_prefetch(A.ecc_in[2] + b16 * 10 * 33 + (size_t)tid * 82, 0, 1);
    __builtin_prefetch(A.dcc_in[0] + b16 * 10 * 33 + (size_t)tid * 82, 0, 1);
  }

  // ---- features: mag, sr, si ----
  const float* spec = A.spec + (size_t)b * 514;
  for (int f = tid; f < 257; f += NT) {
    const float sr = spec[f * 2 + 0];
    const float si = spec[f * 2 + 1];
    s_fh[f] = sqrtf(sr * sr + si * si + 1e-12f);
    s_fh[257 + f] = sr;
    s_fh[514 + f] = si;
  }
  __syncthreads();
  // ---- ERB analysis: copy 65 + (3x192)@(64x192)^T ----
  for (int i = tid; i < 3 * 65; i += NT) { const int c = i / 65, f = i % 65; s_fe[c * 129 + f] = s_fh[c * 257 + f]; }
  block_gemm_f16(s_fh + 65, 257, false, P.erb_bm, 192, s_g1, 64, 3, 64, 192);
  __syncthreads();
  for (int i = tid; i < 3 * 64; i += NT) { const int c = i / 64, n = i % 64; s_fe[c * 129 + 65 + n] = s_g1[c * 64 + n]; }
  __syncthreads();
  // ---- SFE with halo: s_sfe[ch*133 + 2 + f] = fe[c][f + k - 1] ----
  for (int i = tid; i < 9 * 133; i += NT) {
    const int ch = i / 133, fx = i % 133 - 2;
    const int c = ch / 3, k = ch % 3;
    const int ff = fx + k - 1;
    s_sfe[i] = (fx >= 0 && fx < 129 && ff >= 0 && ff < 129) ? s_fe[c * 129 + ff] : 0.f;
  }
  __syncthreads();
  // ---- en0: conv(9->16, k5, s2) + bn + prelu (branchless taps via halo) ----
  if (tid < 64) {  // zero e0 halo cells
    const int co = tid >> 2, k = tid & 3;
    s_e0[co * 69 + ((k < 2) ? k : 65 + k)] = 0.f;
  }
  {
    const float a0 = P.en0.a[0];
    for (int i = tid; i < 16 * 65; i += NT) {
      const int co = i / 65, mo = i % 65;
      float acc = P.en0.b[co];
      for (int ci = 0; ci < 9; ++ci) {
        const float* src = s_sfe + ci * 133 + 2 * mo;   // + (2+ff) with ff = 2mo+d-2
#pragma unroll
        for (int d = 0; d < 5; ++d) acc += src[d] * P.en0.w[(co * 9 + ci) * 5 + d];
      }
      acc = (acc - P.en0.bn.m[co]) * (P.en0.bn.g[co] * rsqrtf(P.en0.bn.v[co] + 1e-5f)) + P.en0.bn.b[co];
      s_e0[co * 69 + 2 + mo] = acc >= 0.f ? acc : a0 * acc;
    }
  }
  __syncthreads();
  // ---- en1: grouped conv(16->16, g2, k5, s2) ----
  {
    const float a0 = P.en1.a[0];
    for (int i = tid; i < 16 * 33; i += NT) {
      const int co = i / 33, mo = i % 33;
      const int g = co >> 3;
      float acc = P.en1.b[co];
      for (int ci = 0; ci < 8; ++ci) {
        const float* src = s_e0 + (g * 8 + ci) * 69 + 2 * mo;  // halo handles edges
#pragma unroll
        for (int d = 0; d < 5; ++d) acc += src[d] * P.en1.w[(co * 8 + ci) * 5 + d];
      }
      acc = (acc - P.en1.bn.m[co]) * (P.en1.bn.g[co] * rsqrtf(P.en1.bn.v[co] + 1e-5f)) + P.en1.bn.b[co];
      s_e1[i] = acc >= 0.f ? acc : a0 * acc;
    }
  }
  __syncthreads();

  // ---- encoder GT blocks ----
  gt_block(P.en2, s_e1, s_xa, A.ecc_in[0] + b16 * 2 * 33, A.ecc_out[0] + b16 * 2 * 33,
           A.etc_in[0] + b16, A.etc_out[0] + b16, 1, 0,
           s_h, s_c0, s_cd, s_y, s_y2, s_g1, s_zt, s_gg, s_htra, s_at);
  for (int i = tid; i < 528; i += NT) s_e2[i] = s_xa[i];
  gt_block(P.en3, s_xa, s_xb, A.ecc_in[1] + b16 * 4 * 33, A.ecc_out[1] + b16 * 4 * 33,
           A.etc_in[1] + b16, A.etc_out[1] + b16, 2, 0,
           s_h, s_c0, s_cd, s_y, s_y2, s_g1, s_zt, s_gg, s_htra, s_at);
  for (int i = tid; i < 528; i += NT) s_e3[i] = s_xb[i];
  gt_block(P.en4, s_xb, s_xa, A.ecc_in[2] + b16 * 10 * 33, A.ecc_out[2] + b16 * 10 * 33,
           A.etc_in[2] + b16, A.etc_out[2] + b16, 5, 0,
           s_h, s_c0, s_cd, s_y, s_y2, s_g1, s_zt, s_gg, s_htra, s_at);
  for (int i = tid; i < 528; i += NT) s_e4[i] = s_xa[i];
  __syncthreads();

  // ---- dual-path GRNN x2 ----
  dpgrnn(P.dp1, s_xa, s_xb, A.ic_in[0] + (size_t)b * 528, A.ic_out[0] + (size_t)b * 528,
         s_i16, s_g1, s_g2, s_io, s_y16, s_hc, s_red);
  dpgrnn(P.dp2, s_xb, s_xa, A.ic_in[1] + (size_t)b * 528, A.ic_out[1] + (size_t)b * 528,
         s_i16, s_g1, s_g2, s_io, s_y16, s_hc, s_red);

  // ---- decoder GT blocks (skip adds + deconv weight flips) ----
  for (int i = tid; i < 528; i += NT) s_xa[i] += s_e4[i];
  __syncthreads();
  gt_block(P.de0, s_xa, s_xb, A.dcc_in[0] + b16 * 10 * 33, A.dcc_out[0] + b16 * 10 * 33,
           A.dtc_in[0] + b16, A.dtc_out[0] + b16, 5, 1,
           s_h, s_c0, s_cd, s_y, s_y2, s_g1, s_zt, s_gg, s_htra, s_at);
  for (int i = tid; i < 528; i += NT) s_xb[i] += s_e3[i];
  __syncthreads();
  gt_block(P.de1, s_xb, s_xa, A.dcc_in[1] + b16 * 4 * 33, A.dcc_out[1] + b16 * 4 * 33,
           A.dtc_in[1] + b16, A.dtc_out[1] + b16, 2, 1,
           s_h, s_c0, s_cd, s_y, s_y2, s_g1, s_zt, s_gg, s_htra, s_at);
  for (int i = tid; i < 528; i += NT) s_xa[i] += s_e2[i];
  __syncthreads();
  gt_block(P.de2, s_xa, s_xb, A.dcc_in[2] + b16 * 2 * 33, A.dcc_out[2] + b16 * 2 * 33,
           A.dtc_in[2] + b16, A.dtc_out[2] + b16, 1, 1,
           s_h, s_c0, s_cd, s_y, s_y2, s_g1, s_zt, s_gg, s_htra, s_at);
  for (int i = tid; i < 528; i += NT) s_xb[i] += s_e1[i];
  __syncthreads();

  // ---- de3: grouped transposed conv (lhs_dilation 2, flipped k5) -> 16 x 65 ----
  {
    const float a0 = P.de3.a[0];
    for (int i = tid; i < 16 * 65; i += NT) {
      const int co = i / 65, mo = i % 65;
      const int g = co >> 3;
      float acc = P.de3.b[co];
      for (int ci = 0; ci < 8; ++ci) {
#pragma unroll
        for (int d = 0; d < 5; ++d) {
          const int ii = mo + d - 2;
          if (ii >= 0 && ii <= 64 && (ii & 1) == 0)
            acc += s_xb[(g * 8 + ci) * 33 + (ii >> 1)] * P.de3.w[(co * 8 + ci) * 5 + (4 - d)];
        }
      }
      acc = (acc - P.de3.bn.m[co]) * (P.de3.bn.g[co] * rsqrtf(P.de3.bn.v[co] + 1e-5f)) + P.de3.bn.b[co];
      s_x65[i] = acc >= 0.f ? acc : a0 * acc;
    }
  }
  __syncthreads();
  for (int i = tid; i < 16 * 65; i += NT) {
    const int co = i / 65, mo = i % 65;
    s_x65[i] += s_e0[co * 69 + 2 + mo];
  }
  __syncthreads();
  // ---- de4: transposed conv (16->2) + bn + tanh -> 2 x 129 ----
  for (int i = tid; i < 2 * 129; i += NT) {
    const int co = i / 129, mo = i % 129;
    float acc = P.de4.b[co];
    for (int ci = 0; ci < 16; ++ci) {
#pragma unroll
      for (int d = 0; d < 5; ++d) {
        const int ii = mo + d - 2;
        if (ii >= 0 && ii <= 128 && (ii & 1) == 0)
          acc += s_x65[ci * 65 + (ii >> 1)] * P.de4.w[(co * 16 + ci) * 5 + (4 - d)];
      }
    }
    acc = (acc - P.de4.bn.m[co]) * (P.de4.bn.g[co] * rsqrtf(P.de4.bn.v[co] + 1e-5f)) + P.de4.bn.b[co];
    s_d4[i] = tanhf(acc);
  }
  __syncthreads();
  // ---- ERB synthesis: (2x64) @ (192x64)^T ----
  block_gemm_f16(s_d4 + 65, 129, false, P.erb_bs, 64, s_mE, 192, 2, 192, 64);
  __syncthreads();
  // ---- complex mask + output ----
  for (int f = tid; f < 257; f += NT) {
    const float m0 = (f < 65) ? s_d4[f] : s_mE[f - 65];
    const float m1 = (f < 65) ? s_d4[129 + f] : s_mE[192 + f - 65];
    const float sr = s_fh[257 + f];
    const float si = s_fh[514 + f];
    A.o_spec[((size_t)b * 257 + f) * 2 + 0] = sr * m0 - si * m1;
    A.o_spec[((size_t)b * 257 + f) * 2 + 1] = si * m0 + sr * m1;
  }
}

// ---------------------------------------------------------------------------
// Host side: d_in[0..14] are the tensors in setup_inputs() order; the params
// pytree is assumed flattened in nested sorted-key order (jax pytree dicts):
// top: de0,de1,de2,de3,de4,dp1,dp2,en0,en1,en2,en3,en4,erb_bm_w,erb_bs_w.
// ---------------------------------------------------------------------------
extern "C" void kernel_launch(void* const* d_in, const int* in_sizes, int n_in,
                              void* d_out, int out_size, void* d_ws, size_t ws_size,
                              hipStream_t stream)
{
  (void)n_in; (void)out_size; (void)d_ws; (void)ws_size;
  const int B = in_sizes[0] / (257 * 2);

  KArgs A;
  A.spec = (const float*)d_in[0];
  A.ecc_in[0] = (const float*)d_in[1];
  A.ecc_in[1] = (const float*)d_in[2];
  A.ecc_in[2] = (const float*)d_in[3];
  A.dcc_in[0] = (const float*)d_in[4];
  A.dcc_in[1] = (const float*)d_in[5];
  A.dcc_in[2] = (const float*)d_in[6];
  A.etc_in[0] = (const float*)d_in[7];
  A.etc_in[1] = (const float*)d_in[8];
  A.etc_in[2] = (const float*)d_in[9];
  A.dtc_in[0] = (const float*)d_in[10];
  A.dtc_in[1] = (const float*)d_in[11];
  A.dtc_in[2] = (const float*)d_in[12];
  A.ic_in[0] = (const float*)d_in[13];
  A.ic_in[1] = (const float*)d_in[14];

  int idx = 15;
  auto nx = [&]() { return (const float*)d_in[idx++]; };
  auto ldBN = [&](BNp& o) { o.b = nx(); o.g = nx(); o.m = nx(); o.v = nx(); };
  auto ldGT = [&](GTp& o) {
    o.a1 = nx(); o.a2 = nx();
    ldBN(o.bn1); ldBN(o.bn2); ldBN(o.bn3);
    o.dw_b = nx(); o.dw_w = nx();
    o.fc_b = nx(); o.fc_w = nx();
    o.gWh = nx(); o.gWi = nx(); o.gbh = nx(); o.gbi = nx();
    o.pc1_b = nx(); o.pc1_w = nx(); o.pc2_b = nx(); o.pc2_w = nx();
  };
  auto ldCB = [&](CBp& o, bool has_a) {
    o.a = has_a ? nx() : nullptr;
    o.b = nx(); ldBN(o.bn); o.w = nx();
  };
  auto ldDP = [&](DPp& o) {
    for (int g = 0; g < 2; ++g) {
      o.iaWh[g][0] = nx(); o.iaWh[g][1] = nx();
      o.iaWi[g][0] = nx(); o.iaWi[g][1] = nx();
      o.iabh[g][0] = nx(); o.iabh[g][1] = nx();
      o.iabi[g][0] = nx(); o.iabi[g][1] = nx();
    }
    for (int g = 0; g < 2; ++g) {
      o.ieWh[g] = nx(); o.ieWi[g] = nx(); o.iebh[g] = nx(); o.iebi[g] = nx();
    }
    o.inter_fc_b = nx(); o.inter_fc_w = nx(); o.inter_ln_b = nx(); o.inter_ln_g = nx();
    o.intra_fc_b = nx(); o.intra_fc_w = nx(); o.intra_ln_b = nx(); o.intra_ln_g = nx();
  };
  ldGT(A.P.de0); ldGT(A.P.de1); ldGT(A.P.de2);
  ldCB(A.P.de3, true); ldCB(A.P.de4, false);
  ldDP(A.P.dp1); ldDP(A.P.dp2);
  ldCB(A.P.en0, true); ldCB(A.P.en1, true);
  ldGT(A.P.en2); ldGT(A.P.en3); ldGT(A.P.en4);
  A.P.erb_bm = nx(); A.P.erb_bs = nx();

  // output slices in reference return order
  float* o = (float*)d_out;
  A.o_spec = o;      o += (size_t)B * 514;
  A.ecc_out[0] = o;  o += (size_t)B * 16 * 2 * 33;
  A.ecc_out[1] = o;  o += (size_t)B * 16 * 4 * 33;
  A.ecc_out[2] = o;  o += (size_t)B * 16 * 10 * 33;
  A.dcc_out[0] = o;  o += (size_t)B * 16 * 10 * 33;
  A.dcc_out[1] = o;  o += (size_t)B * 16 * 4 * 33;
  A.dcc_out[2] = o;  o += (size_t)B * 16 * 2 * 33;
  for (int i = 0; i < 3; ++i) { A.etc_out[i] = o; o += (size_t)B * 16; }
  for (int i = 0; i < 3; ++i) { A.dtc_out[i] = o; o += (size_t)B * 16; }
  A.ic_out[0] = o;   o += (size_t)B * 528;
  A.ic_out[1] = o;

  gtcrn_mega_kernel<<<dim3((unsigned)B), dim3(NT_BLK), 0, stream>>>(A);
}
